// DualFusionLayer_40544491274781
// MI455X (gfx1250) — matrix operand order repaired and checked by reading, not compiled, not run on hardware
//
#include <hip/hip_runtime.h>
#include <hip/hip_bf16.h>
#include <stdint.h>

// Problem constants (match reference)
static constexpr int    M_ROWS = 100000;
static constexpr int    N_ROWS = 200000;
static constexpr int    CH     = 128;     // C
static constexpr int    CH2    = 256;     // 2C

typedef __bf16  v16bf __attribute__((ext_vector_type(16)));
typedef __bf16  v8bf  __attribute__((ext_vector_type(8)));
typedef __bf16  v4bf  __attribute__((ext_vector_type(4)));
typedef float   v8f   __attribute__((ext_vector_type(8)));
typedef float   v4f   __attribute__((ext_vector_type(4)));

static __device__ __forceinline__ void atomAddF(float* p, float v) {
  __hip_atomic_fetch_add(p, v, __ATOMIC_RELAXED, __HIP_MEMORY_SCOPE_AGENT);
}

// ---------------------------------------------------------------------------
// Zero the accumulator/counter region of the workspace (128-bit stores).
// ---------------------------------------------------------------------------
__global__ void zero_ws(v4f* __restrict__ p, long n4) {
  long i = blockIdx.x * (long)blockDim.x + threadIdx.x;
  long stride = (long)gridDim.x * blockDim.x;
  v4f z = {};
  for (; i < n4; i += stride) p[i] = z;
}

// ---------------------------------------------------------------------------
// One wave per edge: gather both endpoint feature rows (float4/lane) and
// scatter with hardware f32 atomics (L2-resident; x_v+x_f fit in 192MB L2).
// ---------------------------------------------------------------------------
__global__ __launch_bounds__(256) void scatter_edges(
    const float* __restrict__ xv, const float* __restrict__ xf,
    const int* __restrict__ er, const int* __restrict__ ec,
    float* __restrict__ sum_fv, float* __restrict__ cnt_v,
    float* __restrict__ sum_vf, float* __restrict__ cnt_f, int E) {
  int e    = (blockIdx.x * blockDim.x + threadIdx.x) >> 5;
  int lane = threadIdx.x & 31;
  if (e >= E) return;
  int r = er[e];                     // vertex index
  int c = ec[e];                     // face index
  int co = lane * 4;
  v4f f4 = *(const v4f*)(xf + (size_t)c * CH + co);
  v4f v4 = *(const v4f*)(xv + (size_t)r * CH + co);
  float* df = sum_fv + (size_t)r * CH + co;   // faces -> vertex
  float* dv = sum_vf + (size_t)c * CH + co;   // vertices -> face
#pragma unroll
  for (int i = 0; i < 4; ++i) {
    atomAddF(df + i, f4[i]);
    atomAddF(dv + i, v4[i]);
  }
  if (lane == 0) {
    atomAddF(cnt_v + r, 1.0f);
    atomAddF(cnt_f + c, 1.0f);
  }
}

// ---------------------------------------------------------------------------
// Build bf16 H = concat(x, sum / max(cnt,1)) row-major [rows, 2C].
// ---------------------------------------------------------------------------
__global__ void pack_h(const float* __restrict__ x, const float* __restrict__ sum,
                       const float* __restrict__ cnt, __bf16* __restrict__ h, int rows) {
  long total = (long)rows * (CH / 4);
  long i = blockIdx.x * (long)blockDim.x + threadIdx.x;
  long stride = (long)gridDim.x * blockDim.x;
  for (; i < total; i += stride) {
    long m = i / (CH / 4);
    int  c = (int)(i % (CH / 4)) * 4;
    v4f xs = *(const v4f*)(x + m * CH + c);
    v4f ss = *(const v4f*)(sum + m * CH + c);
    float inv = 1.0f / fmaxf(cnt[m], 1.0f);
    v4bf a, b;
#pragma unroll
    for (int j = 0; j < 4; ++j) {
      a[j] = (__bf16)xs[j];
      b[j] = (__bf16)(ss[j] * inv);
    }
    *(v4bf*)(h + m * CH2 + c) = a;
    *(v4bf*)(h + m * CH2 + CH + c) = b;
  }
}

// Small f32 -> bf16 weight conversion.
__global__ void cvt_bf16(const float* __restrict__ s, __bf16* __restrict__ d, int n) {
  int i = blockIdx.x * blockDim.x + threadIdx.x;
  if (i < n) d[i] = (__bf16)s[i];
}

// ---------------------------------------------------------------------------
// Fused 2-layer MLP on 16-row tiles.
//   out = leaky( leaky(H @ W1^T + b1) @ W2^T + b2 ),  H:[rows,256] bf16
// W1:[128,256] bf16 row-major, W2:[128,128] bf16 row-major (so B[k][n]=W[n][k]
// -> each B lane fragment is 16 contiguous bf16 of one W row).
// 8 waves/block, wave w owns output columns 16w..16w+15. Inter-layer
// activation lives in a 4KB LDS tile; only final f32 result hits HBM.
// ---------------------------------------------------------------------------
__global__ __launch_bounds__(256) void fused_mlp(
    const __bf16* __restrict__ H,  const __bf16* __restrict__ W1,
    const float*  __restrict__ b1, const __bf16* __restrict__ W2,
    const float*  __restrict__ b2, float* __restrict__ out, int rows) {
  __shared__ __bf16 h1s[16 * CH];

  const int wv   = threadIdx.x >> 5;
  const int lane = threadIdx.x & 31;
  const int half = lane >> 4;   // 0: lanes 0-15, 1: lanes 16-31
  const int lx   = lane & 15;
  const long tile_m = (long)blockIdx.x * 16;
  if (tile_m >= rows) return;

  // --- GEMM1: [16 x 256] x [256 x 16] per wave, K-steps of 32 ---
  const __bf16* arow = H  + (tile_m + lx) * CH2;                 // A row m
  const __bf16* w1r  = W1 + (size_t)(wv * 16 + lx) * CH2 + half * 16;  // B col n
  const __bf16* w2r  = W2 + (size_t)(wv * 16 + lx) * CH  + half * 16;
  __builtin_prefetch(w2r, 0, 1);   // warm L2/L0 for layer 2 weights

  v8f acc = {};
#pragma unroll
  for (int kk = 0; kk < 8; ++kk) {
    const int k0 = kk * 32;
    // A fragment: 16 contiguous halves at K = k0 + 16*half (matches B).
    v8bf alo = *(const v8bf*)(arow + k0 + half * 16);
    v8bf ahi = *(const v8bf*)(arow + k0 + half * 16 + 8);
    v16bf a = __builtin_shufflevector(alo, ahi, 0,1,2,3,4,5,6,7,8,9,10,11,12,13,14,15);
    // B fragment: 16 contiguous halves of W1 row n at K = k0 + 16*half.
    v16bf b = *(const v16bf*)(w1r + k0);
    acc = __builtin_amdgcn_wmma_f32_16x16x32_bf16(
        false, a, false, b, (short)0, acc, false, false);
  }

  // bias + leaky ReLU, stage bf16 activation tile in LDS.
  const float bias1 = b1[wv * 16 + lx];
#pragma unroll
  for (int r = 0; r < 8; ++r) {
    float v = acc[r] + bias1;
    v = v > 0.0f ? v : 0.2f * v;
    h1s[(r + half * 8) * CH + wv * 16 + lx] = (__bf16)v;   // [m][n]
  }
  __syncthreads();

  // --- GEMM2: [16 x 128] x [128 x 16] per wave, A from LDS ---
  v8f acc2 = {};
#pragma unroll
  for (int kk = 0; kk < 4; ++kk) {
    const int k0 = kk * 32;
    const __bf16* ar = h1s + lx * CH;
    v8bf alo = *(const v8bf*)(ar + k0 + half * 16);
    v8bf ahi = *(const v8bf*)(ar + k0 + half * 16 + 8);
    v16bf a = __builtin_shufflevector(alo, ahi, 0,1,2,3,4,5,6,7,8,9,10,11,12,13,14,15);
    v16bf b = *(const v16bf*)(w2r + k0);
    acc2 = __builtin_amdgcn_wmma_f32_16x16x32_bf16(
        false, a, false, b, (short)0, acc2, false, false);
  }

  const float bias2 = b2[wv * 16 + lx];
#pragma unroll
  for (int r = 0; r < 8; ++r) {
    float v = acc2[r] + bias2;
    v = v > 0.0f ? v : 0.2f * v;
    out[(tile_m + r + half * 8) * CH + wv * 16 + lx] = v;
  }
}

// ---------------------------------------------------------------------------
extern "C" void kernel_launch(void* const* d_in, const int* in_sizes, int n_in,
                              void* d_out, int out_size, void* d_ws, size_t ws_size,
                              hipStream_t stream) {
  const float* x_v = (const float*)d_in[0];
  const float* x_f = (const float*)d_in[1];
  const int*   er  = (const int*)d_in[2];
  const int*   ec  = (const int*)d_in[3];
  const float* Wv1 = (const float*)d_in[4];
  const float* bv1 = (const float*)d_in[5];
  const float* Wv2 = (const float*)d_in[6];
  const float* bv2 = (const float*)d_in[7];
  const float* Wf1 = (const float*)d_in[8];
  const float* bf1 = (const float*)d_in[9];
  const float* Wf2 = (const float*)d_in[10];
  const float* bf2 = (const float*)d_in[11];
  const int E = in_sizes[2];

  // Workspace carve-up (256B aligned slices).
  char* ws = (char*)d_ws;
  size_t off = 0;
  auto carve = [&](size_t bytes) -> size_t {
    size_t p = off;
    off = (off + bytes + 255) & ~(size_t)255;
    return p;
  };
  const size_t o_sum_fv = carve((size_t)M_ROWS * CH * 4);
  const size_t o_sum_vf = carve((size_t)N_ROWS * CH * 4);
  const size_t o_cnt_v  = carve((size_t)M_ROWS * 4);
  const size_t o_cnt_f  = carve((size_t)N_ROWS * 4);
  const size_t zero_end = off;                       // everything above zeroed
  const size_t o_hv     = carve((size_t)M_ROWS * CH2 * 2);
  const size_t o_hf     = carve((size_t)N_ROWS * CH2 * 2);
  const size_t o_wv1    = carve((size_t)CH * CH2 * 2);
  const size_t o_wv2    = carve((size_t)CH * CH * 2);
  const size_t o_wf1    = carve((size_t)CH * CH2 * 2);
  const size_t o_wf2    = carve((size_t)CH * CH * 2);
  (void)ws_size;

  float*  sum_fv = (float*)(ws + o_sum_fv);
  float*  sum_vf = (float*)(ws + o_sum_vf);
  float*  cnt_v  = (float*)(ws + o_cnt_v);
  float*  cnt_f  = (float*)(ws + o_cnt_f);
  __bf16* hv     = (__bf16*)(ws + o_hv);
  __bf16* hf     = (__bf16*)(ws + o_hf);
  __bf16* wbv1   = (__bf16*)(ws + o_wv1);
  __bf16* wbv2   = (__bf16*)(ws + o_wv2);
  __bf16* wbf1   = (__bf16*)(ws + o_wf1);
  __bf16* wbf2   = (__bf16*)(ws + o_wf2);

  // 1) zero accumulators + counters
  zero_ws<<<4096, 256, 0, stream>>>((v4f*)ws, (long)(zero_end / 16));

  // 2) edge scatter (one wave per edge)
  scatter_edges<<<(E + 7) / 8, 256, 0, stream>>>(
      x_v, x_f, er, ec, sum_fv, cnt_v, sum_vf, cnt_f, E);

  // 3) pack bf16 activations + weights
  pack_h<<<2048, 256, 0, stream>>>(x_v, sum_fv, cnt_v, hv, M_ROWS);
  pack_h<<<4096, 256, 0, stream>>>(x_f, sum_vf, cnt_f, hf, N_ROWS);
  cvt_bf16<<<(CH * CH2 + 255) / 256, 256, 0, stream>>>(Wv1, wbv1, CH * CH2);
  cvt_bf16<<<(CH * CH  + 255) / 256, 256, 0, stream>>>(Wv2, wbv2, CH * CH);
  cvt_bf16<<<(CH * CH2 + 255) / 256, 256, 0, stream>>>(Wf1, wbf1, CH * CH2);
  cvt_bf16<<<(CH * CH  + 255) / 256, 256, 0, stream>>>(Wf2, wbf2, CH * CH);

  // 4) fused WMMA 2-layer MLPs (M and N are exact multiples of 16)
  float* out_v = (float*)d_out;
  float* out_f = out_v + (size_t)M_ROWS * CH;
  fused_mlp<<<M_ROWS / 16, 256, 0, stream>>>(hv, wbv1, bv1, wbv2, bv2, out_v, M_ROWS);
  fused_mlp<<<N_ROWS / 16, 256, 0, stream>>>(hf, wbf1, bf1, wbf2, bf2, out_f, N_ROWS);
}
